// NeuralNetworkOnAEV_91156385890903
// MI455X (gfx1250) — matrix operand order repaired and checked by reading, not compiled, not run on hardware
//
#include <hip/hip_runtime.h>
#include <stdint.h>

// ---------------------------------------------------------------------------
// ANI per-species MLP, fused, bf16 WMMA (f32 accumulate) for gfx1250 / MI455X
//   aev [B,64,384] f32 -> L0(384->160) gauss -> L1(160->128) gauss
//   -> L2(128->96) gauss -> L3(96->1) -> sum over atoms -> out[B]
// B-fragments: n-major K-pair-contiguous LDS layout -> 2x ds_load_b128 each.
// A-fragments: bf16 staging, 2x ds_load_b128 each, cached in regs per layer.
// fp32->bf16 hot-path packing via single v_perm_b32 per dword.
// ---------------------------------------------------------------------------

typedef __attribute__((ext_vector_type(16))) __bf16 v16bf;
typedef __attribute__((ext_vector_type(8)))  float  v8f;

#define NATOMS      64
#define D0          384
#define D1          160
#define D2          128
#define D3          96
#define WAVES       8
#define ROWS_PER_WG (16 * WAVES)      // 128 batch rows per workgroup
#define ACT_STRIDE  160               // staging row stride (bf16 elements)

// n-major weight buffer: lds_w[n*KPAD + kp], KPAD multiple of 4, != 0 mod 64
#define KPAD0 100                     // L0 chunk: 96 k-pairs (192 K rows)
#define KPAD1 84                      // L1: 80 k-pairs
#define KPAD2 68                      // L2: 64 k-pairs

#define SMEM_W_B    (160 * KPAD0 * 4)              // 64000 B (max layer need)
#define SMEM_BIAS_B 640
#define SMEM_W3_B   512
#define SMEM_ACT_B  (WAVES * 16 * ACT_STRIDE * 2)  // 40960 B
#define SMEM_TOTAL  (SMEM_W_B + SMEM_BIAS_B + SMEM_W3_B + SMEM_ACT_B)

#define NLOG2E (-1.442695040888963f)

union FragBF { v16bf v; uint32_t u[8]; uint4 q[2]; };

// RNE fp32->bf16 (used only in the once-per-WG weight conversion)
__device__ __forceinline__ uint32_t f2bf_rne(float x) {
    uint32_t u = __float_as_uint(x);
    u += 0x7FFFu + ((u >> 16) & 1u);
    return u >> 16;
}
__device__ __forceinline__ uint32_t pack2bf_rne(float lo, float hi) {
    return f2bf_rne(lo) | (f2bf_rne(hi) << 16);
}

// Truncating fp32x2 -> packed bf16x2: one v_perm_b32.
// result bytes {hi[3],hi[2],lo[3],lo[2]} -> selector 0x07060302 (S0=hi, S1=lo)
__device__ __forceinline__ uint32_t pack2bf_fast(float lo, float hi) {
    return __builtin_amdgcn_perm(__float_as_uint(hi), __float_as_uint(lo),
                                 0x07060302u);
}

__device__ __forceinline__ float bf2f(uint16_t h) {
    return __uint_as_float(((uint32_t)h) << 16);
}

// gauss(x) = exp(-x^2) = exp2(x*x * -log2(e))  -> 2 VALU + v_exp_f32
__device__ __forceinline__ float gaussf(float x) {
    return __builtin_amdgcn_exp2f(x * x * NLOG2E);
}

// Cooperative fp32 -> bf16 pair-packed weight load, n-major:
// lds_w[n*kpad + kp] = {bf16 W[2kp+1,n] , bf16 W[2kp,n]}
// idx sweeps kp-major so global reads stay n-contiguous (coalesced).
__device__ __forceinline__ void coop_load_w(const float* __restrict__ W,
                                            uint32_t* lds_w, int kpairs, int N,
                                            int kpad, int tid) {
    int total = kpairs * N;
    for (int idx = tid; idx < total; idx += 256) {
        int kp = idx / N;
        int n  = idx - kp * N;
        float w0 = W[(2 * kp)     * N + n];
        float w1 = W[(2 * kp + 1) * N + n];
        lds_w[n * kpad + kp] = pack2bf_rne(w0, w1);
    }
}

// ISA A-frag layout (16-bit, 16x32): VGPR v holds K = (v>>2)*16 + half*8 + 2*(v&3) (+1)
// ISA B-frag layout (16-bit, 32x16): VGPR v holds K = (v>>2)*16 + half*8 + 2*(v&3) (+1), N = lane&15
template <int K, int N, int KPAD>
__device__ __forceinline__ void hidden_layer(const uint32_t* __restrict__ lds_w,
                                             const float* __restrict__ lds_b,
                                             uint16_t* __restrict__ act,
                                             int r, int hf) {
    constexpr int KT = K / 32;
    constexpr int NT = N / 16;
    const uint32_t* act32 = (const uint32_t*)act;

    // cache all A-fragments for this wave's 16 rows (reads precede act writes)
    FragBF a[KT];
#pragma unroll
    for (int t = 0; t < KT; ++t) {
        int base = r * (ACT_STRIDE / 2) + t * 16 + hf * 4;
        a[t].q[0] = *(const uint4*)(act32 + base);
        a[t].q[1] = *(const uint4*)(act32 + base + 8);
    }
#pragma unroll
    for (int j = 0; j < NT; ++j) {
        float bj = lds_b[j * 16 + r];
        v8f c;
#pragma unroll
        for (int i = 0; i < 8; ++i) c[i] = bj;
        int n = j * 16 + r;
#pragma unroll
        for (int t = 0; t < KT; ++t) {
            FragBF bm;
            const uint4* bp = (const uint4*)(lds_w + n * KPAD + t * 16 + hf * 4);
            bm.q[0] = bp[0];
            bm.q[1] = bp[2];
            c = __builtin_amdgcn_wmma_f32_16x16x32_bf16(
                    false, a[t].v, false, bm.v, (short)0, c, false, false);
        }
        // gaussian activation; truncating bf16 store (d16_hi pattern)
#pragma unroll
        for (int i = 0; i < 8; ++i) {
            uint32_t gb = __float_as_uint(gaussf(c[i]));
            act[(i + 8 * hf) * ACT_STRIDE + n] = (uint16_t)(gb >> 16);
        }
    }
}

__global__ void zero_out_kernel(float* __restrict__ out, int n) {
    int i = blockIdx.x * blockDim.x + threadIdx.x;
    if (i < n) out[i] = 0.0f;
}

__global__ __launch_bounds__(256) void ani_fused_kernel(
    const float* __restrict__ aev, const int* __restrict__ species,
    const float* __restrict__ W0g, const float* __restrict__ b0g,
    const float* __restrict__ W1g, const float* __restrict__ b1g,
    const float* __restrict__ W2g, const float* __restrict__ b2g,
    const float* __restrict__ W3g, const float* __restrict__ b3g,
    float* __restrict__ out) {
    const int tid  = threadIdx.x;
    const int wave = tid >> 5;
    const int lane = tid & 31;
    const int hf   = lane >> 4;   // lane-half (ISA A/B/C layout split)
    const int r    = lane & 15;
    const int atom = blockIdx.y;
    const int row0 = blockIdx.x * ROWS_PER_WG + wave * 16;

    extern __shared__ char smem[];
    uint32_t* lds_w  = (uint32_t*)smem;
    float*    lds_b  = (float*)(smem + SMEM_W_B);
    float*    lds_w3 = (float*)(smem + SMEM_W_B + SMEM_BIAS_B);
    uint16_t* act    = (uint16_t*)(smem + SMEM_W_B + SMEM_BIAS_B + SMEM_W3_B) +
                       wave * 16 * ACT_STRIDE;

    const int s = species[atom];
    const float* W0s = W0g + (size_t)s * D0 * D1;
    const float* W1s = W1g + (size_t)s * D1 * D2;
    const float* W2s = W2g + (size_t)s * D2 * D3;
    const float* W3s = W3g + (size_t)s * D3;

    // ------------------- layer 0: [16x384] x [384x160] ---------------------
    // A from global (fp32 -> bf16 via v_perm_b32); all 10 N-tile accumulators
    // stay live so AEV is streamed from HBM exactly once. W0 in two K=192
    // chunks to keep the LDS weight buffer small (3 WGs/WGP possible).
    v8f c0[10];
    const float* arow = aev + ((size_t)(row0 + r) * NATOMS + atom) * D0;

    for (int ck = 0; ck < 2; ++ck) {
        __syncthreads();
        coop_load_w(W0s + (size_t)(ck * 192) * D1, lds_w, 96, D1, KPAD0, tid);
        if (ck == 0)
            for (int idx = tid; idx < D1; idx += 256)
                lds_b[idx] = b0g[s * D1 + idx];
        __syncthreads();
        if (ck == 0) {
#pragma unroll
            for (int j = 0; j < 10; ++j) {
                float bj = lds_b[j * 16 + r];
#pragma unroll
                for (int i = 0; i < 8; ++i) c0[j][i] = bj;
            }
        }
#pragma unroll
        for (int tl = 0; tl < 6; ++tl) {
            // lane needs k in [t*32 + hf*8, +8) and [t*32 + 16 + hf*8, +8)
            const float* src = arow + (ck * 6 + tl) * 32 + hf * 8;
            float4 f0 = ((const float4*)src)[0];
            float4 f1 = ((const float4*)src)[1];
            float4 f2 = ((const float4*)(src + 16))[0];
            float4 f3 = ((const float4*)(src + 16))[1];
            FragBF a;
            a.u[0] = pack2bf_fast(f0.x, f0.y); a.u[1] = pack2bf_fast(f0.z, f0.w);
            a.u[2] = pack2bf_fast(f1.x, f1.y); a.u[3] = pack2bf_fast(f1.z, f1.w);
            a.u[4] = pack2bf_fast(f2.x, f2.y); a.u[5] = pack2bf_fast(f2.z, f2.w);
            a.u[6] = pack2bf_fast(f3.x, f3.y); a.u[7] = pack2bf_fast(f3.z, f3.w);
            int kpb = tl * 16 + hf * 4;
#pragma unroll
            for (int j = 0; j < 10; ++j) {
                FragBF bm;
                const uint4* bp =
                    (const uint4*)(lds_w + (j * 16 + r) * KPAD0 + kpb);
                bm.q[0] = bp[0];
                bm.q[1] = bp[2];
                c0[j] = __builtin_amdgcn_wmma_f32_16x16x32_bf16(
                            false, a.v, false, bm.v, (short)0, c0[j], false, false);
            }
        }
    }
    // gaussian + truncating bf16 stage
#pragma unroll
    for (int j = 0; j < 10; ++j) {
        int n = j * 16 + r;
#pragma unroll
        for (int i = 0; i < 8; ++i) {
            uint32_t gb = __float_as_uint(gaussf(c0[j][i]));
            act[(i + 8 * hf) * ACT_STRIDE + n] = (uint16_t)(gb >> 16);
        }
    }

    // ------------------- layer 1: 160 -> 128 --------------------------------
    __syncthreads();
    coop_load_w(W1s, lds_w, 80, D2, KPAD1, tid);
    for (int idx = tid; idx < D2; idx += 256) lds_b[idx] = b1g[s * D2 + idx];
    __syncthreads();
    hidden_layer<160, 128, KPAD1>(lds_w, lds_b, act, r, hf);

    // ------------------- layer 2: 128 -> 96 ---------------------------------
    __syncthreads();
    coop_load_w(W2s, lds_w, 64, D3, KPAD2, tid);
    for (int idx = tid; idx < D3; idx += 256) lds_b[idx] = b2g[s * D3 + idx];
    __syncthreads();
    hidden_layer<128, 96, KPAD2>(lds_w, lds_b, act, r, hf);

    // ------------------- layer 3: 96 -> 1, sum over atoms -------------------
    __syncthreads();
    for (int idx = tid; idx < D3; idx += 256) lds_w3[idx] = W3s[idx];
    if (tid == 0) lds_w3[D3] = b3g[s];
    __syncthreads();
    if (hf == 0) {   // lanes 0..15, one batch row each; 96 MACs is VALU noise
        float sum = 0.0f;
#pragma unroll
        for (int k = 0; k < D3; ++k)
            sum += bf2f(act[r * ACT_STRIDE + k]) * lds_w3[k];
        atomicAdd(&out[row0 + r], sum + lds_w3[D3]);
    }
}

extern "C" void kernel_launch(void* const* d_in, const int* in_sizes, int n_in,
                              void* d_out, int out_size, void* d_ws, size_t ws_size,
                              hipStream_t stream) {
    const float* aev     = (const float*)d_in[0];
    const int*   species = (const int*)d_in[1];
    const float* W0g     = (const float*)d_in[2];
    const float* b0g     = (const float*)d_in[3];
    const float* W1g     = (const float*)d_in[4];
    const float* b1g     = (const float*)d_in[5];
    const float* W2g     = (const float*)d_in[6];
    const float* b2g     = (const float*)d_in[7];
    const float* W3g     = (const float*)d_in[8];
    const float* b3g     = (const float*)d_in[9];
    float* out = (float*)d_out;

    int Bn = in_sizes[0] / (NATOMS * D0);   // 2048
    if (Bn < ROWS_PER_WG) Bn = ROWS_PER_WG;

    // d_out is poisoned; energies are accumulated with atomics, so zero first.
    zero_out_kernel<<<(out_size + 255) / 256, 256, 0, stream>>>(out, out_size);

    dim3 grid(Bn / ROWS_PER_WG, NATOMS);    // (16, 64)
    ani_fused_kernel<<<grid, 256, SMEM_TOTAL, stream>>>(
        aev, species, W0g, b0g, W1g, b1g, W2g, b2g, W3g, b3g, out);
}
